// STFTTransformerBase_44186623541469
// MI455X (gfx1250) — compile-verified, matrix operand order checked by reading, not run on hardware
//
#include <hip/hip_runtime.h>
#include <math.h>
#include <stdint.h>

// ---------------- problem constants ----------------
#define BATCH      32
#define NCH        8
#define T_LEN      4128
#define FFT_WIN    64
#define FFT_STRIDE 32
#define NFREQ      33
#define NFREQ_PAD  64
#define TFRAMES    128          // (4128-64)/32+1
#define D_MODEL    256
#define NHEADS     8
#define HDIM       32
#define NLAYERS    4
#define D_FF       1024
#define S_TOK      1025         // 128*8 + 1 (CLS)
#define ROWS_ALL   (BATCH * S_TOK)   // 32800, divisible by 16 (not by 64)
#define LN_EPS     1e-5f

typedef __attribute__((ext_vector_type(16))) _Float16 v16h;
typedef __attribute__((ext_vector_type(8)))  float    v8f;

union Frag16 {
    v16h     v;
    uint32_t u[8];
    _Float16 h[16];
};

// K index of the first element of dword j for 16-bit A/B fragments
// (ISA 7.12.2: VGPR j<4 -> K = 2j + 8*half, VGPR j>=4 -> K = 16 + 2(j-4) + 8*half)
__device__ __forceinline__ int k0_of(int j, int half) {
    return ((j & 3) << 1) + ((j >> 2) << 4) + (half << 3);
}

// Load a 16x32 f16 fragment from a row-major buffer (global or LDS);
// consecutive-K pairs are contiguous so each dword is one load.
__device__ __forceinline__ v16h load_frag_rowpair(const _Float16* __restrict__ src, int ld) {
    const int lane = threadIdx.x & 31;
    const int m = lane & 15, hf = lane >> 4;
    Frag16 f;
    const _Float16* row = src + (size_t)m * ld;
#pragma unroll
    for (int j = 0; j < 8; ++j)
        f.u[j] = *(const uint32_t*)(row + k0_of(j, hf));
    return f.v;
}

// ---- CDNA5 async global->LDS staging (ASYNCcnt-tracked DMA, ISA 15.18.3) ----
// ldsOff: byte offset in LDS (VDST operand); ga: 64-bit global address.
__device__ __forceinline__ void async_b128(uint32_t ldsOff, uint64_t ga) {
    asm volatile("global_load_async_to_lds_b128 %0, %1, off"
                 :: "v"(ldsOff), "v"(ga) : "memory");
}

// Copy a 64-row x 32-half tile (row stride ld halves) into LDS at ldsOff.
// 128 threads x 2 chunks of 16B each = 64*64B tile. Rows clamped to maxRow.
__device__ __forceinline__ void async_copy_tile(const _Float16* __restrict__ gbase,
                                                int ld, int rowOff, int maxRow,
                                                uint32_t ldsOff) {
    const int tid = threadIdx.y * 32 + threadIdx.x;   // 0..127
#pragma unroll
    for (int i = 0; i < 2; ++i) {
        const int e = tid + i * 128;                  // chunk id 0..255
        const int r = e >> 2, c = e & 3;              // row 0..63, 16B chunk 0..3
        int gr = rowOff + r; if (gr > maxRow) gr = maxRow;
        const uint64_t ga = (uint64_t)(uintptr_t)(gbase + (size_t)gr * ld) +
                            (uint64_t)(c * 16);
        async_b128(ldsOff + (uint32_t)(r * 64 + c * 16), ga);
    }
}

// ---------------- WMMA GEMM with double-buffered async LDS staging ----------------
//   C[M,N] = A16[M,K] @ W16[N,K]^T + bias
// Block tile 64x64, K-step 32. 4 waves; wave w owns rows [w*16, w*16+16) x 64 cols
// (one A fragment reused across 4 accumulating WMMAs).
// OUT_MODE: 0 = store f32, 1 = add into f32 (residual), 2 = store f16
template <int OUT_MODE, bool GELU_ACT>
__global__ __launch_bounds__(128) void gemm_wmma(const _Float16* __restrict__ A,
                                                 const _Float16* __restrict__ W,
                                                 const float* __restrict__ bias,
                                                 float* __restrict__ outF,
                                                 _Float16* __restrict__ outH,
                                                 int M, int N, int K) {
    __shared__ __align__(16) _Float16 sA[2][64 * 32];
    __shared__ __align__(16) _Float16 sB[2][64 * 32];

    const int lane = threadIdx.x;          // 0..31
    const int wid  = threadIdx.y;          // 0..3
    const int rowBase = blockIdx.x * 64;
    const int colBase = blockIdx.y * 64;

    const uint32_t aOff[2] = { (uint32_t)(uintptr_t)&sA[0][0],
                               (uint32_t)(uintptr_t)&sA[1][0] };
    const uint32_t bOff[2] = { (uint32_t)(uintptr_t)&sB[0][0],
                               (uint32_t)(uintptr_t)&sB[1][0] };

    // prologue: stage first K tile
    async_copy_tile(A, K, rowBase, M - 1, aOff[0]);
    async_copy_tile(W, K, colBase, N - 1, bOff[0]);

    v8f acc[4] = {};
    const int nK = K / 32;
    for (int t = 0; t < nK; ++t) {
        const int buf = t & 1;
        if (t + 1 < nK) {   // prefetch next tile into the other buffer
            async_copy_tile(A + (size_t)(t + 1) * 32, K, rowBase, M - 1, aOff[buf ^ 1]);
            async_copy_tile(W + (size_t)(t + 1) * 32, K, colBase, N - 1, bOff[buf ^ 1]);
            // 4 newer per-thread async ops outstanding; <=4 left => current tile landed
            asm volatile("s_wait_asynccnt 4" ::: "memory");
        } else {
            asm volatile("s_wait_asynccnt 0" ::: "memory");
        }
        __syncthreads();

        const v16h a = load_frag_rowpair(&sA[buf][wid * 16 * 32], 32);
#pragma unroll
        for (int g = 0; g < 4; ++g) {
            const v16h b = load_frag_rowpair(&sB[buf][g * 16 * 32], 32);
            acc[g] = __builtin_amdgcn_wmma_f32_16x16x32_f16(false, a, false, b,
                                                            (short)0, acc[g], false, false);
        }
        __syncthreads();    // all waves done reading before this buffer is re-filled
    }

    const int n = lane & 15, hf = lane >> 4;
#pragma unroll
    for (int g = 0; g < 4; ++g) {
        const int col = colBase + g * 16 + n;
        const float bv = bias ? bias[col] : 0.0f;
#pragma unroll
        for (int i = 0; i < 8; ++i) {
            const int row = rowBase + wid * 16 + i + (hf << 3);
            if (row >= M) continue;
            float v = acc[g][i] + bv;
            if (GELU_ACT) v = 0.5f * v * (1.0f + erff(v * 0.70710678118f));
            const size_t idx = (size_t)row * N + col;
            if (OUT_MODE == 0)      outF[idx] = v;
            else if (OUT_MODE == 1) outF[idx] += v;
            else                    outH[idx] = (_Float16)v;
        }
    }
}

// ---------------- attention mask bias (computed analytically) ----------------
__device__ __forceinline__ float mask_bias(int q, int k) {
    if (q >= S_TOK || k >= S_TOK) return -1e30f;   // tile padding
    if (q == S_TOK - 1) return 0.0f;               // CLS attends to everything
    if (k == S_TOK - 1) return -1e9f;              // signal tokens cannot see CLS
    return ((k >> 3) > (q >> 3)) ? -1e9f : 0.0f;   // block-causal over time ids
}

// ---------------- flash-style attention, one wave per 16-query tile per (b,h) ----------------
// qkv layout: [b][token][768] f16  (q: 0..255, k: 256..511, v: 512..767)
__global__ void attention_kernel(const _Float16* __restrict__ qkv,
                                 _Float16* __restrict__ o16) {
    const int qt   = blockIdx.x;        // query tile, 0..64
    const int h    = blockIdx.y;        // head
    const int b    = blockIdx.z;        // batch
    const int lane = threadIdx.x;       // 0..31
    const int lidx = lane & 15, lhf = lane >> 4;

    __shared__ _Float16 pld[16][34];    // probs staged to remap D-layout -> A-layout

    const size_t seqBase = (size_t)b * S_TOK * (3 * D_MODEL);
    const int    qbase   = qt * 16;

    // Q fragment (A-layout, K = hd = 32)
    Frag16 qa;
    {
        int tok = qbase + lidx; if (tok >= S_TOK) tok = S_TOK - 1;
        const _Float16* qrow = qkv + seqBase + (size_t)tok * (3 * D_MODEL) + h * HDIM;
#pragma unroll
        for (int j = 0; j < 8; ++j)
            qa.u[j] = *(const uint32_t*)(qrow + k0_of(j, lhf));
    }

    v8f oA = {}, oB = {};
    float mrun[8], lrun[8];
#pragma unroll
    for (int i = 0; i < 8; ++i) { mrun[i] = -3.0e38f; lrun[i] = 0.0f; }

    const float scale = 0.17677669529663687f;   // 1/sqrt(32)

    for (int kb = 0; kb < S_TOK; kb += 32) {
        // ---- scores = Q @ K^T over this 32-key chunk (two 16-key WMMAs) ----
        v8f s0, s1;
        {
            int t0 = kb + lidx;      if (t0 >= S_TOK) t0 = S_TOK - 1;
            int t1 = kb + 16 + lidx; if (t1 >= S_TOK) t1 = S_TOK - 1;
            const _Float16* kp0 = qkv + seqBase + (size_t)t0 * (3 * D_MODEL) + D_MODEL + h * HDIM;
            const _Float16* kp1 = qkv + seqBase + (size_t)t1 * (3 * D_MODEL) + D_MODEL + h * HDIM;
            Frag16 kb0, kb1;
#pragma unroll
            for (int j = 0; j < 8; ++j) {
                const int k0 = k0_of(j, lhf);
                kb0.u[j] = *(const uint32_t*)(kp0 + k0);
                kb1.u[j] = *(const uint32_t*)(kp1 + k0);
            }
            v8f z = {};
            s0 = __builtin_amdgcn_wmma_f32_16x16x32_f16(false, qa.v, false, kb0.v, (short)0, z, false, false);
            s1 = __builtin_amdgcn_wmma_f32_16x16x32_f16(false, qa.v, false, kb1.v, (short)0, z, false, false);
        }

        __syncthreads();   // previous iteration's P reads done before overwrite

        // ---- mask + online softmax (row reductions across 16-lane halves) ----
#pragma unroll
        for (int i = 0; i < 8; ++i) {
            const int qtok = qbase + i + (lhf << 3);
            float v0 = s0[i] * scale + mask_bias(qtok, kb + lidx);
            float v1 = s1[i] * scale + mask_bias(qtok, kb + 16 + lidx);

            float m2 = fmaxf(v0, v1);
#pragma unroll
            for (int msk = 1; msk < 16; msk <<= 1)
                m2 = fmaxf(m2, __shfl_xor(m2, msk, 32));

            const float mnew = fmaxf(mrun[i], m2);
            const float corr = __expf(mrun[i] - mnew);
            lrun[i] *= corr;
            oA[i] *= corr;
            oB[i] *= corr;

            const float p0 = __expf(v0 - mnew);
            const float p1 = __expf(v1 - mnew);
            float ls = p0 + p1;
#pragma unroll
            for (int msk = 1; msk < 16; msk <<= 1)
                ls += __shfl_xor(ls, msk, 32);
            lrun[i] += ls;
            mrun[i] = mnew;

            pld[i + (lhf << 3)][lidx]      = (_Float16)p0;
            pld[i + (lhf << 3)][16 + lidx] = (_Float16)p1;
        }
        __syncthreads();

        // ---- P as A-fragment (from LDS), V as B-fragments, O += P @ V ----
        Frag16 pa;
#pragma unroll
        for (int j = 0; j < 8; ++j) {
            const int k0 = k0_of(j, lhf);
            pa.h[2 * j]     = pld[lidx][k0];
            pa.h[2 * j + 1] = pld[lidx][k0 + 1];
        }
        Frag16 vb0, vb1;
#pragma unroll
        for (int e = 0; e < 16; ++e) {
            const int kk = k0_of(e >> 1, lhf) + (e & 1);
            int tok = kb + kk; if (tok >= S_TOK) tok = S_TOK - 1;
            const _Float16* vp = qkv + seqBase + (size_t)tok * (3 * D_MODEL) + 2 * D_MODEL + h * HDIM;
            vb0.h[e] = vp[lidx];
            vb1.h[e] = vp[16 + lidx];
        }
        oA = __builtin_amdgcn_wmma_f32_16x16x32_f16(false, pa.v, false, vb0.v, (short)0, oA, false, false);
        oB = __builtin_amdgcn_wmma_f32_16x16x32_f16(false, pa.v, false, vb1.v, (short)0, oB, false, false);
    }

    // ---- normalize and write O (f16, feeds out-proj GEMM) ----
#pragma unroll
    for (int i = 0; i < 8; ++i) {
        const int tok = qbase + i + (lhf << 3);
        if (tok < S_TOK) {
            const float inv = (lrun[i] > 0.0f) ? (1.0f / lrun[i]) : 0.0f;
            const size_t orow = ((size_t)b * S_TOK + tok) * D_MODEL + h * HDIM;
            o16[orow + lidx]      = (_Float16)(oA[i] * inv);
            o16[orow + 16 + lidx] = (_Float16)(oB[i] * inv);
        }
    }
}

// ---------------- LayerNorm: f32 row -> f16 row (GEMM input) ----------------
__global__ void layernorm_kernel(const float* __restrict__ x,
                                 const float* __restrict__ w,
                                 const float* __restrict__ bsrc,
                                 _Float16* __restrict__ out) {
    const int row = blockIdx.x, t = threadIdx.x;   // blockDim = 256
    __shared__ float red[256];
    const float v = x[(size_t)row * D_MODEL + t];
    red[t] = v; __syncthreads();
    for (int s = 128; s > 0; s >>= 1) { if (t < s) red[t] += red[t + s]; __syncthreads(); }
    const float mu = red[0] * (1.0f / D_MODEL);
    __syncthreads();
    const float d = v - mu;
    red[t] = d * d; __syncthreads();
    for (int s = 128; s > 0; s >>= 1) { if (t < s) red[t] += red[t + s]; __syncthreads(); }
    const float var = red[0] * (1.0f / D_MODEL);
    out[(size_t)row * D_MODEL + t] = (_Float16)(d * rsqrtf(var + LN_EPS) * w[t] + bsrc[t]);
}

// ---------------- STFT front-end ----------------
__global__ void l2norm_kernel(const float* __restrict__ emg, float* __restrict__ scl) {
    const int bc = blockIdx.x, t = threadIdx.x;    // 256 blocks x 256 thr
    __shared__ float red[256];
    float s = 0.0f;
    const float* p = emg + (size_t)bc * T_LEN;
    for (int i = t; i < T_LEN; i += 256) { const float v = p[i]; s += v * v; }
    red[t] = s; __syncthreads();
    for (int k = 128; k > 0; k >>= 1) { if (t < k) red[t] += red[t + k]; __syncthreads(); }
    if (t == 0) scl[bc] = 1.0f / fmaxf(sqrtf(red[0]), 1e-12f);
}

__global__ void stft_kernel(const float* __restrict__ emg,
                            const float* __restrict__ scl,
                            _Float16* __restrict__ spec) {
    const int frame = blockIdx.x & (TFRAMES - 1);
    const int bc    = blockIdx.x >> 7;
    const int t     = threadIdx.x;                 // 64 threads
    __shared__ float sbuf[FFT_WIN];
    const float hann = 0.5f - 0.5f * __cosf(6.28318530718f * t / (float)(FFT_WIN - 1));
    sbuf[t] = emg[(size_t)bc * T_LEN + frame * FFT_STRIDE + t] * scl[bc] * hann;
    __syncthreads();
    const size_t row = ((size_t)bc * TFRAMES + frame) * NFREQ_PAD;
    if (t < NFREQ) {
        float re = 0.0f, im = 0.0f;
        const float w0 = 6.28318530718f * t / (float)FFT_WIN;
        for (int n = 0; n < FFT_WIN; ++n) {
            float sn, cn;
            __sincosf(w0 * n, &sn, &cn);
            re += sbuf[n] * cn;
            im -= sbuf[n] * sn;
        }
        spec[row + t] = (_Float16)log1pf(sqrtf(re * re + im * im));
    } else {
        spec[row + t] = (_Float16)0.0f;            // K padding (proj weights also zero there)
    }
}

// ---------------- token assembly: proj output + channel PE + CLS -> x ----------------
__global__ void assemble_kernel(const float* __restrict__ ptmp,
                                const float* __restrict__ cls,
                                float* __restrict__ x) {
    const int d = threadIdx.x;                     // 256
    const int row = blockIdx.x;                    // b*1025 + tok
    const int b = row / S_TOK, tok = row % S_TOK;
    float v;
    if (tok == S_TOK - 1) {
        v = cls[d];
    } else {
        const int tt = tok >> 3, c = tok & 7;
        v = ptmp[(((size_t)(b * NCH + c) * TFRAMES) + tt) * D_MODEL + d];
        if (d < NCH) {                             // channel positional encoding
            const float theta = 0.78539816339f * (float)c;   // 2*pi*c/8
            const int   f     = d >> 1;
            const float a     = (float)(f + 1) * theta;
            v += (d & 1) ? __cosf(a) : __sinf(a);
        }
    }
    x[(size_t)row * D_MODEL + d] = v;
}

__global__ void cls_kernel(const float* __restrict__ x, float* __restrict__ out) {
    out[blockIdx.x * D_MODEL + threadIdx.x] =
        x[((size_t)blockIdx.x * S_TOK + (S_TOK - 1)) * D_MODEL + threadIdx.x];
}

// ---------------- weight conversion ----------------
__global__ void f32_to_f16_kernel(const float* __restrict__ src, _Float16* __restrict__ dst, int n) {
    const int i = blockIdx.x * 256 + threadIdx.x;
    if (i < n) dst[i] = (_Float16)src[i];
}

__global__ void pad_projw_kernel(const float* __restrict__ src, _Float16* __restrict__ dst) {
    const int i = blockIdx.x * 64 + threadIdx.x;   // 256 blocks x 64 -> [256][64]
    const int d = i >> 6, f = i & 63;
    dst[i] = (f < NFREQ) ? (_Float16)src[d * NFREQ + f] : (_Float16)0.0f;
}

// ---------------- host orchestration ----------------
extern "C" void kernel_launch(void* const* d_in, const int* in_sizes, int n_in,
                              void* d_out, int out_size, void* d_ws, size_t ws_size,
                              hipStream_t stream) {
    (void)in_sizes; (void)n_in; (void)out_size; (void)ws_size;

    const float* emg   = (const float*)d_in[0];
    const float* projw = (const float*)d_in[1];
    const float* projb = (const float*)d_in[2];
    const float* cls   = (const float*)d_in[3];
    const float* inw   = (const float*)d_in[4];
    const float* inb   = (const float*)d_in[5];
    const float* outw  = (const float*)d_in[6];
    const float* outb  = (const float*)d_in[7];
    const float* ln1w  = (const float*)d_in[8];
    const float* ln1b  = (const float*)d_in[9];
    const float* ln2w  = (const float*)d_in[10];
    const float* ln2b  = (const float*)d_in[11];
    const float* f1w   = (const float*)d_in[12];
    const float* f1b   = (const float*)d_in[13];
    const float* f2w   = (const float*)d_in[14];
    const float* f2b   = (const float*)d_in[15];
    float* out = (float*)d_out;

    char* ws = (char*)d_ws;
    size_t off = 0;
    auto take = [&](size_t bytes) {
        char* p = ws + off;
        off = (off + bytes + 255) & ~(size_t)255;
        return p;
    };

    const int N_INW = NLAYERS * 3 * D_MODEL * D_MODEL;   // 786432
    const int N_OUW = NLAYERS * D_MODEL * D_MODEL;       // 262144
    const int N_F1W = NLAYERS * D_FF * D_MODEL;          // 1048576
    const int N_F2W = NLAYERS * D_MODEL * D_FF;          // 1048576
    const int M_SPEC = BATCH * NCH * TFRAMES;            // 32768

    _Float16* w16_in  = (_Float16*)take((size_t)N_INW * 2);
    _Float16* w16_out = (_Float16*)take((size_t)N_OUW * 2);
    _Float16* w16_f1  = (_Float16*)take((size_t)N_F1W * 2);
    _Float16* w16_f2  = (_Float16*)take((size_t)N_F2W * 2);
    _Float16* w16_p   = (_Float16*)take((size_t)D_MODEL * NFREQ_PAD * 2);
    float*    scl     = (float*)   take((size_t)BATCH * NCH * 4);
    _Float16* spec16  = (_Float16*)take((size_t)M_SPEC * NFREQ_PAD * 2);
    float*    x       = (float*)   take((size_t)ROWS_ALL * D_MODEL * 4);
    _Float16* h16     = (_Float16*)take((size_t)ROWS_ALL * D_MODEL * 2);
    _Float16* o16     = (_Float16*)take((size_t)ROWS_ALL * D_MODEL * 2);
    // union: proj tmp (f32) / qkv (f16) / ff activations (f16) — disjoint lifetimes
    char* big = take((size_t)ROWS_ALL * D_FF * 2);
    float*    ptmp  = (float*)big;
    _Float16* qkv16 = (_Float16*)big;
    _Float16* ff16  = (_Float16*)big;

    // --- weight conversion (every call; deterministic) ---
    f32_to_f16_kernel<<<(N_INW + 255) / 256, 256, 0, stream>>>(inw,  w16_in,  N_INW);
    f32_to_f16_kernel<<<(N_OUW + 255) / 256, 256, 0, stream>>>(outw, w16_out, N_OUW);
    f32_to_f16_kernel<<<(N_F1W + 255) / 256, 256, 0, stream>>>(f1w,  w16_f1,  N_F1W);
    f32_to_f16_kernel<<<(N_F2W + 255) / 256, 256, 0, stream>>>(f2w,  w16_f2,  N_F2W);
    pad_projw_kernel<<<D_MODEL, 64, 0, stream>>>(projw, w16_p);

    // --- STFT features ---
    l2norm_kernel<<<BATCH * NCH, 256, 0, stream>>>(emg, scl);
    stft_kernel<<<M_SPEC, 64, 0, stream>>>(emg, scl, spec16);

    // --- input projection (WMMA GEMM, K padded to 64) ---
    dim3 gblk(32, 4);
    const int mt_spec = M_SPEC / 64;                     // 512
    const int mt_all  = (ROWS_ALL + 63) / 64;            // 513
    gemm_wmma<0, false><<<dim3(mt_spec, D_MODEL / 64), gblk, 0, stream>>>(
        spec16, w16_p, projb, ptmp, nullptr, M_SPEC, D_MODEL, NFREQ_PAD);
    assemble_kernel<<<ROWS_ALL, D_MODEL, 0, stream>>>(ptmp, cls, x);

    // --- transformer layers ---
    for (int l = 0; l < NLAYERS; ++l) {
        layernorm_kernel<<<ROWS_ALL, D_MODEL, 0, stream>>>(
            x, ln1w + l * D_MODEL, ln1b + l * D_MODEL, h16);
        gemm_wmma<2, false><<<dim3(mt_all, (3 * D_MODEL) / 64), gblk, 0, stream>>>(
            h16, w16_in + (size_t)l * 3 * D_MODEL * D_MODEL, inb + l * 3 * D_MODEL,
            nullptr, qkv16, ROWS_ALL, 3 * D_MODEL, D_MODEL);
        attention_kernel<<<dim3((S_TOK + 15) / 16, NHEADS, BATCH), 32, 0, stream>>>(qkv16, o16);
        gemm_wmma<1, false><<<dim3(mt_all, D_MODEL / 64), gblk, 0, stream>>>(
            o16, w16_out + (size_t)l * D_MODEL * D_MODEL, outb + l * D_MODEL,
            x, nullptr, ROWS_ALL, D_MODEL, D_MODEL);
        layernorm_kernel<<<ROWS_ALL, D_MODEL, 0, stream>>>(
            x, ln2w + l * D_MODEL, ln2b + l * D_MODEL, h16);
        gemm_wmma<2, true><<<dim3(mt_all, D_FF / 64), gblk, 0, stream>>>(
            h16, w16_f1 + (size_t)l * D_FF * D_MODEL, f1b + l * D_FF,
            nullptr, ff16, ROWS_ALL, D_FF, D_MODEL);
        gemm_wmma<1, false><<<dim3(mt_all, D_MODEL / 64), gblk, 0, stream>>>(
            ff16, w16_f2 + (size_t)l * D_MODEL * D_FF, f2b + l * D_MODEL,
            x, nullptr, ROWS_ALL, D_MODEL, D_FF);
    }

    // --- output: CLS token per batch ---
    cls_kernel<<<BATCH, D_MODEL, 0, stream>>>(x, out);
}
